// VehicleGATNetwork_68478958567729
// MI455X (gfx1250) — matrix operand order repaired and checked by reading, not compiled, not run on hardware
//
#include <hip/hip_runtime.h>
#include <math.h>

#define HIDF   128
#define NHEAD  4
#define DHEAD  32
#define NODEF  15
#define EDGEF  10
#define GDIMF  8
#define NLAYER 3
#define ZCOLS  160   // 4 heads * 32 + strat 32
#define POOLROWS 256

typedef __attribute__((ext_vector_type(16))) _Float16 v16h;
typedef __attribute__((ext_vector_type(8)))  float    v8f;

// ---------------- monotone float<->uint for atomic max ----------------
__device__ __forceinline__ unsigned flipf(float f) {
    unsigned u = __float_as_uint(f);
    return (u & 0x80000000u) ? ~u : (u | 0x80000000u);
}
__device__ __forceinline__ float unflipf(unsigned u) {
    return __uint_as_float((u & 0x80000000u) ? (u & 0x7fffffffu) : ~u);
}

// ---------------- WMMA GEMM: C[MxNC] = act(A[MxK] @ B[KxNC] + bias) ----------------
// A,B,C row-major f32.  One wave per block computes a 16x32 output tile:
// one A fragment is reused by two v_wmma (two adjacent 16x16 N-tiles).
// Requires M % 16 == 0, NC % 32 == 0, K % 32 == 0 (all exact for this net).
template<int K, int NC, bool BIAS, bool RELU>
__global__ __launch_bounds__(32) void wmma_gemm(
    const float* __restrict__ A, const float* __restrict__ B,
    const float* __restrict__ bias, float* __restrict__ C)
{
    const int lane = threadIdx.x & 31;
    const int m0 = blockIdx.x * 16;
    const int n0 = blockIdx.y * 32;
    const int r    = lane & 15;   // A row / B,C column within tile
    const int half = lane >> 4;

    const float* Arow = A + (size_t)(m0 + r) * K;
    v8f acc0 = {}, acc1 = {};

#pragma unroll
    for (int kb = 0; kb < K; kb += 32) {
        // A fragment (16x32 f16): lanes 0-15 hold K {0..7,16..23}, lanes 16-31 {8..15,24..31}
        v16h a;
        {
            const float* p0 = Arow + kb + half * 8;
            const float* p1 = Arow + kb + 16 + half * 8;
#pragma unroll
            for (int i = 0; i < 8; ++i) a[i] = (_Float16)p0[i];
#pragma unroll
            for (int i = 0; i < 8; ++i) a[8 + i] = (_Float16)p1[i];
        }
        // B fragments (32x16 f16 each): lane col = r, lanes 0-15 hold K {0..7,16..23}, lanes 16-31 {8..15,24..31}
        v16h b0, b1;
        {
            const float* q = B + (size_t)(kb + half * 8) * NC + n0 + r;
#pragma unroll
            for (int i = 0; i < 16; ++i) b0[i] = (_Float16)q[(i < 8 ? i : i + 8) * NC];
#pragma unroll
            for (int i = 0; i < 16; ++i) b1[i] = (_Float16)q[(i < 8 ? i : i + 8) * NC + 16];
        }
        acc0 = __builtin_amdgcn_wmma_f32_16x16x32_f16(false, a, false, b0,
                                                      (short)0, acc0, false, false);
        acc1 = __builtin_amdgcn_wmma_f32_16x16x32_f16(false, a, false, b1,
                                                      (short)0, acc1, false, false);
    }

    const float bv0 = BIAS ? bias[n0 + r] : 0.0f;
    const float bv1 = BIAS ? bias[n0 + 16 + r] : 0.0f;
    float* Crow = C + (size_t)(m0 + half * 8) * NC + n0 + r;   // C/D: VGPR j -> row half*8+j
#pragma unroll
    for (int j = 0; j < 8; ++j) {
        float v0 = acc0[j] + bv0;
        float v1 = acc1[j] + bv1;
        if (RELU) { v0 = fmaxf(v0, 0.0f); v1 = fmaxf(v1, 0.0f); }
        Crow[(size_t)j * NC]      = v0;
        Crow[(size_t)j * NC + 16] = v1;
    }
}

// ---------------- tiny precompute kernels ----------------
__global__ void k_weh(const float* __restrict__ lin_edge, const float* __restrict__ att_edge,
                      float* __restrict__ weh) {
    int t = blockIdx.x * blockDim.x + threadIdx.x;        // l*40 + k*4 + h
    if (t >= NLAYER * EDGEF * NHEAD) return;
    int l = t / (EDGEF * NHEAD);
    int k = (t / NHEAD) % EDGEF;
    int h = t % NHEAD;
    const float* le = lin_edge + (size_t)l * EDGEF * HIDF + (size_t)k * HIDF + h * DHEAD;
    const float* ae = att_edge + (size_t)l * NHEAD * DHEAD + h * DHEAD;
    float s = 0.0f;
    for (int d = 0; d < DHEAD; ++d) s += le[d] * ae[d];
    weh[t] = s;
}

__global__ void k_wcat(const float* __restrict__ hw1, const float* __restrict__ hb1,
                       const float* __restrict__ sw1, const float* __restrict__ sb1,
                       float* __restrict__ Wcat, float* __restrict__ bcat) {
    int t = blockIdx.x * blockDim.x + threadIdx.x;
    if (t < HIDF * ZCOLS) {
        int f = t / ZCOLS, c = t % ZCOLS;
        float v;
        if (c < 128) { int k = c >> 5, o = c & 31; v = hw1[(size_t)k * HIDF * 32 + (size_t)f * 32 + o]; }
        else          v = sw1[(size_t)f * 32 + (c - 128)];
        Wcat[t] = v;
    }
    if (t < ZCOLS)
        bcat[t] = (t < 128) ? hb1[(t >> 5) * 32 + (t & 31)] : sb1[t - 128];
}

// ---------------- self-loop edge-attr mean ----------------
__global__ void k_self_acc(const int* __restrict__ ei, const float* __restrict__ eattr,
                           float* __restrict__ cnt, float* __restrict__ easum, int E) {
    int e = blockIdx.x * blockDim.x + threadIdx.x;
    if (e >= E) return;
    int d = ei[E + e];
    atomicAdd(&cnt[d], 1.0f);
    const float* row = eattr + (size_t)e * EDGEF;
    for (int k = 0; k < EDGEF; ++k) atomicAdd(&easum[(size_t)d * EDGEF + k], row[k]);
}
__global__ void k_self_div(float* __restrict__ eaself, const float* __restrict__ cnt, int N) {
    int t = blockIdx.x * blockDim.x + threadIdx.x;
    if (t >= N * EDGEF) return;
    eaself[t] = eaself[t] / fmaxf(cnt[t / EDGEF], 1.0f);
}

// ---------------- node encoder (K=15, VALU) ----------------
__global__ __launch_bounds__(128) void k_encoder(const float* __restrict__ x,
                                                 const float* __restrict__ w,
                                                 const float* __restrict__ b,
                                                 float* __restrict__ h, int N) {
    int n = blockIdx.x, j = threadIdx.x;
    const float* xr = x + (size_t)n * NODEF;
    float s = b[j];
    for (int k = 0; k < NODEF; ++k) s += xr[k] * w[k * HIDF + j];
    h[(size_t)n * HIDF + j] = fmaxf(s, 0.0f);
}

// ---------------- attention coefficients a_s, a_d ----------------
__global__ void k_attcoef(const float* __restrict__ hp, const float* __restrict__ att_s,
                          const float* __restrict__ att_d, float* __restrict__ a_s,
                          float* __restrict__ a_d, int N) {
    int t = blockIdx.x * blockDim.x + threadIdx.x;
    if (t >= N * NHEAD) return;
    int n = t >> 2, h = t & 3;
    const float* row = hp + (size_t)n * HIDF + h * DHEAD;
    const float* as = att_s + h * DHEAD;
    const float* ad = att_d + h * DHEAD;
    float ss = 0.0f, sd = 0.0f;
    for (int d = 0; d < DHEAD; ++d) { float v = row[d]; ss += v * as[d]; sd += v * ad[d]; }
    a_s[t] = ss; a_d[t] = sd;
}

// ---------------- alpha + segment max ----------------
__global__ void k_alpha(const int* __restrict__ ei, const float* __restrict__ eattr,
                        const float* __restrict__ eaself, const float* __restrict__ weh,
                        const float* __restrict__ a_s, const float* __restrict__ a_d,
                        float* __restrict__ wbuf, unsigned* __restrict__ mflip, int E, int N) {
    int e = blockIdx.x * blockDim.x + threadIdx.x;
    if (e >= E + N) return;
    int s, d; const float* ea;
    if (e < E) { s = ei[e]; d = ei[E + e]; ea = eattr + (size_t)e * EDGEF; }
    else       { s = d = e - E;            ea = eaself + (size_t)s * EDGEF; }
    float eav[EDGEF];
#pragma unroll
    for (int k = 0; k < EDGEF; ++k) eav[k] = ea[k];
#pragma unroll
    for (int h = 0; h < NHEAD; ++h) {
        float aeh = 0.0f;
#pragma unroll
        for (int k = 0; k < EDGEF; ++k) aeh += eav[k] * weh[k * NHEAD + h];
        float al = a_s[s * NHEAD + h] + a_d[d * NHEAD + h] + aeh;
        al = (al > 0.0f) ? al : 0.2f * al;                 // leaky_relu(0.2)
        wbuf[(size_t)e * NHEAD + h] = al;
        atomicMax(&mflip[d * NHEAD + h], flipf(al));
    }
}

// ---------------- exp + segment sum ----------------
__global__ void k_expsum(const int* __restrict__ ei, float* __restrict__ wbuf,
                         const unsigned* __restrict__ mflip, float* __restrict__ denom,
                         int E, int N) {
    int e = blockIdx.x * blockDim.x + threadIdx.x;
    if (e >= E + N) return;
    int d = (e < E) ? ei[E + e] : (e - E);
#pragma unroll
    for (int h = 0; h < NHEAD; ++h) {
        float m = unflipf(mflip[d * NHEAD + h]);
        float w = __expf(wbuf[(size_t)e * NHEAD + h] - m);
        wbuf[(size_t)e * NHEAD + h] = w;
        atomicAdd(&denom[d * NHEAD + h], w);
    }
}

// ---------------- weighted scatter-aggregate ----------------
__global__ __launch_bounds__(128) void k_aggregate(const int* __restrict__ ei,
                                                   const float* __restrict__ wbuf,
                                                   const float* __restrict__ denom,
                                                   const float* __restrict__ hp,
                                                   float* __restrict__ outagg, int E, int N) {
    int e = blockIdx.x;
    int j = threadIdx.x;
    int s, d;
    if (e < E) { s = ei[e]; d = ei[E + e]; } else { s = d = e - E; }
    int h = j >> 5;
    float att = wbuf[(size_t)e * NHEAD + h] / denom[d * NHEAD + h];
    atomicAdd(&outagg[(size_t)d * HIDF + j], att * hp[(size_t)s * HIDF + j]);
}

// ---------------- residual + layernorm ----------------
__global__ __launch_bounds__(128) void k_ln(const float* __restrict__ outagg,
                                            const float* __restrict__ bias,
                                            const float* __restrict__ g,
                                            const float* __restrict__ b,
                                            float* __restrict__ h, int N) {
    __shared__ float red[HIDF];
    int n = blockIdx.x, j = threadIdx.x;
    float t = h[(size_t)n * HIDF + j] + outagg[(size_t)n * HIDF + j] + bias[j];
    red[j] = t; __syncthreads();
    for (int s = 64; s > 0; s >>= 1) { if (j < s) red[j] += red[j + s]; __syncthreads(); }
    float mu = red[0] / HIDF;
    __syncthreads();
    float dv = t - mu;
    red[j] = dv * dv; __syncthreads();
    for (int s = 64; s > 0; s >>= 1) { if (j < s) red[j] += red[j + s]; __syncthreads(); }
    float inv = rsqrtf(red[0] / HIDF + 1e-5f);
    h[(size_t)n * HIDF + j] = dv * inv * g[j] + b[j];
}

// ---------------- pooling partial reduce ----------------
__global__ __launch_bounds__(128) void k_poolred(const float* __restrict__ h,
                                                 float* __restrict__ psum,
                                                 unsigned* __restrict__ pmax, int N) {
    int j = threadIdx.x;
    int r0 = blockIdx.x * POOLROWS;
    int r1 = r0 + POOLROWS; if (r1 > N) r1 = N;
    float s = 0.0f, m = -3.4e38f;
    for (int r = r0; r < r1; ++r) {
        float v = h[(size_t)r * HIDF + j];
        s += v; m = fmaxf(m, v);
    }
    atomicAdd(&psum[j], s);
    atomicMax(&pmax[j], flipf(m));
}

// ---------------- pooled MLP + global head (single block) ----------------
__global__ __launch_bounds__(128) void k_global(const float* __restrict__ psum,
                                                const unsigned* __restrict__ pmax,
                                                const float* __restrict__ u,
                                                const float* __restrict__ pool_w,
                                                const float* __restrict__ pool_b,
                                                const float* __restrict__ pln_g,
                                                const float* __restrict__ pln_b,
                                                const float* __restrict__ gw1,
                                                const float* __restrict__ gb1,
                                                const float* __restrict__ gw2,
                                                const float* __restrict__ gb2,
                                                float* __restrict__ out_global, int N) {
    __shared__ float pooled[3 * HIDF + GDIMF];
    __shared__ float g[GDIMF];
    __shared__ float s1[32];
    int t = threadIdx.x;
    if (t < HIDF) {
        float sum = psum[t];
        pooled[t]            = sum / (float)N;
        pooled[HIDF + t]     = unflipf(pmax[t]);
        pooled[2 * HIDF + t] = sum;
    }
    if (t < GDIMF) pooled[3 * HIDF + t] = u[t];
    __syncthreads();
    if (t < GDIMF) {
        float s = pool_b[t];
        for (int i = 0; i < 3 * HIDF + GDIMF; ++i) s += pooled[i] * pool_w[i * GDIMF + t];
        g[t] = fmaxf(s, 0.0f);
    }
    __syncthreads();
    if (t == 0) {
        float mu = 0.0f; for (int i = 0; i < GDIMF; ++i) mu += g[i]; mu /= GDIMF;
        float var = 0.0f; for (int i = 0; i < GDIMF; ++i) { float d = g[i] - mu; var += d * d; } var /= GDIMF;
        float inv = rsqrtf(var + 1e-5f);
        for (int i = 0; i < GDIMF; ++i) g[i] = (g[i] - mu) * inv * pln_g[i] + pln_b[i];
    }
    __syncthreads();
    if (t < 32) {
        float s = gb1[t];
        for (int i = 0; i < GDIMF; ++i) s += g[i] * gw1[i * 32 + t];
        s1[t] = fmaxf(s, 0.0f);
    }
    __syncthreads();
    if (t < 4) {
        float s = gb2[t];
        for (int o = 0; o < 32; ++o) s += s1[o] * gw2[o * 4 + t];
        out_global[t] = tanhf(s);
    }
}

// ---------------- per-node heads from fused projection (warp per node) ----------------
__global__ __launch_bounds__(256) void k_heads(const float* __restrict__ zcat,
                                               const float* __restrict__ hw2,
                                               const float* __restrict__ hb2,
                                               const float* __restrict__ sw2,
                                               const float* __restrict__ sb2,
                                               float* __restrict__ out, int N) {
    int warp = threadIdx.x >> 5;
    int lane = threadIdx.x & 31;
    int n = blockIdx.x * 8 + warp;
    if (n >= N) return;
    const float* z = zcat + (size_t)n * ZCOLS;
    float y[4];
#pragma unroll
    for (int k = 0; k < 4; ++k) {
        float p = z[k * 32 + lane] * hw2[k * 32 + lane];
        for (int off = 16; off > 0; off >>= 1) p += __shfl_xor(p, off, 32);
        y[k] = p + hb2[k];
    }
    float zs = z[128 + lane];
    float lg[5];
#pragma unroll
    for (int j = 0; j < 5; ++j) {
        float p = zs * sw2[lane * 5 + j];
        for (int off = 16; off > 0; off >>= 1) p += __shfl_xor(p, off, 32);
        lg[j] = p + sb2[j];
    }
    if (lane == 0) {
        out[n]                      = tanhf(y[0]);
        out[(size_t)N + n]          = 1.0f / (1.0f + __expf(-y[1]));
        out[2 * (size_t)N + n]      = 1.0f / (1.0f + __expf(-y[2]));
        out[3 * (size_t)N + n]      = 1.0f / (1.0f + __expf(-y[3]));
        float mx = lg[0];
        for (int j = 1; j < 5; ++j) mx = fmaxf(mx, lg[j]);
        float se = 0.0f, ex[5];
        for (int j = 0; j < 5; ++j) { ex[j] = __expf(lg[j] - mx); se += ex[j]; }
        for (int j = 0; j < 5; ++j) out[4 * (size_t)N + (size_t)n * 5 + j] = ex[j] / se;
    }
}

// ---------------- host launch ----------------
extern "C" void kernel_launch(void* const* d_in, const int* in_sizes, int n_in,
                              void* d_out, int out_size, void* d_ws, size_t ws_size,
                              hipStream_t stream) {
    const float* x        = (const float*)d_in[0];
    const int*   ei       = (const int*)  d_in[1];
    const float* eattr    = (const float*)d_in[2];
    const float* u        = (const float*)d_in[3];
    const float* enc_w    = (const float*)d_in[4];
    const float* enc_b    = (const float*)d_in[5];
    const float* lin_w    = (const float*)d_in[6];
    const float* att_src  = (const float*)d_in[7];
    const float* att_dst  = (const float*)d_in[8];
    const float* lin_edge = (const float*)d_in[9];
    const float* att_edge = (const float*)d_in[10];
    const float* gat_bias = (const float*)d_in[11];
    const float* ln_g     = (const float*)d_in[12];
    const float* ln_b     = (const float*)d_in[13];
    const float* pool_w   = (const float*)d_in[14];
    const float* pool_b   = (const float*)d_in[15];
    const float* pln_g    = (const float*)d_in[16];
    const float* pln_b    = (const float*)d_in[17];
    const float* hw1      = (const float*)d_in[18];
    const float* hb1      = (const float*)d_in[19];
    const float* hw2      = (const float*)d_in[20];
    const float* hb2      = (const float*)d_in[21];
    const float* sw1      = (const float*)d_in[22];
    const float* sb1      = (const float*)d_in[23];
    const float* sw2      = (const float*)d_in[24];
    const float* sb2      = (const float*)d_in[25];
    const float* gw1      = (const float*)d_in[26];
    const float* gb1      = (const float*)d_in[27];
    const float* gw2      = (const float*)d_in[28];
    const float* gb2      = (const float*)d_in[29];

    const int N = in_sizes[0] / NODEF;
    const int E = in_sizes[1] / 2;
    const int TOT = E + N;

    // workspace layout (floats)
    float* ws = (float*)d_ws;
    size_t off = 0;
    float*    h      = ws + off; off += (size_t)N * HIDF;
    float*    hp     = ws + off; off += (size_t)N * HIDF;
    float*    outagg = ws + off; off += (size_t)N * HIDF;
    float*    wbuf   = ws + off; off += (size_t)TOT * NHEAD;
    float*    a_s    = ws + off; off += (size_t)N * NHEAD;
    float*    a_d    = ws + off; off += (size_t)N * NHEAD;
    unsigned* mflip  = (unsigned*)(ws + off); off += (size_t)N * NHEAD;
    float*    denom  = ws + off; off += (size_t)N * NHEAD;
    float*    cnt    = ws + off; off += (size_t)N;
    float*    eaself = ws + off; off += (size_t)N * EDGEF;
    float*    weh    = ws + off; off += NLAYER * EDGEF * NHEAD;
    float*    Wcat   = ws + off; off += HIDF * ZCOLS;
    float*    bcat   = ws + off; off += ZCOLS;
    float*    psum   = ws + off; off += HIDF;
    unsigned* pmax   = (unsigned*)(ws + off); off += HIDF;
    float*    zcat   = hp;  // aliases hp (+ spills into outagg); both free post-layers

    float* out = (float*)d_out;

    hipMemsetAsync(cnt,    0, (size_t)N * sizeof(float), stream);
    hipMemsetAsync(eaself, 0, (size_t)N * EDGEF * sizeof(float), stream);
    hipMemsetAsync(psum,   0, HIDF * sizeof(float), stream);
    hipMemsetAsync(pmax,   0, HIDF * sizeof(unsigned), stream);

    k_self_acc<<<(E + 255) / 256, 256, 0, stream>>>(ei, eattr, cnt, eaself, E);
    k_self_div<<<(N * EDGEF + 255) / 256, 256, 0, stream>>>(eaself, cnt, N);
    k_weh<<<1, 128, 0, stream>>>(lin_edge, att_edge, weh);
    k_wcat<<<(HIDF * ZCOLS + 255) / 256, 256, 0, stream>>>(hw1, hb1, sw1, sb1, Wcat, bcat);
    k_encoder<<<N, 128, 0, stream>>>(x, enc_w, enc_b, h, N);

    const int Mt = N / 16;   // N = 50000 is exactly 3125 * 16
    for (int l = 0; l < NLAYER; ++l) {
        dim3 g1(Mt, HIDF / 32);
        wmma_gemm<HIDF, HIDF, false, false><<<g1, 32, 0, stream>>>(
            h, lin_w + (size_t)l * HIDF * HIDF, nullptr, hp);
        k_attcoef<<<(N * NHEAD + 255) / 256, 256, 0, stream>>>(
            hp, att_src + (size_t)l * NHEAD * DHEAD, att_dst + (size_t)l * NHEAD * DHEAD,
            a_s, a_d, N);
        hipMemsetAsync(mflip,  0, (size_t)N * NHEAD * sizeof(unsigned), stream);
        hipMemsetAsync(denom,  0, (size_t)N * NHEAD * sizeof(float), stream);
        hipMemsetAsync(outagg, 0, (size_t)N * HIDF * sizeof(float), stream);
        k_alpha<<<(TOT + 255) / 256, 256, 0, stream>>>(ei, eattr, eaself,
            weh + l * EDGEF * NHEAD, a_s, a_d, wbuf, mflip, E, N);
        k_expsum<<<(TOT + 255) / 256, 256, 0, stream>>>(ei, wbuf, mflip, denom, E, N);
        k_aggregate<<<TOT, 128, 0, stream>>>(ei, wbuf, denom, hp, outagg, E, N);
        k_ln<<<N, 128, 0, stream>>>(outagg, gat_bias + (size_t)l * HIDF,
                                    ln_g + (size_t)l * HIDF, ln_b + (size_t)l * HIDF, h, N);
    }

    k_poolred<<<(N + POOLROWS - 1) / POOLROWS, 128, 0, stream>>>(h, psum, pmax, N);
    k_global<<<1, 128, 0, stream>>>(psum, pmax, u, pool_w, pool_b, pln_g, pln_b,
                                    gw1, gb1, gw2, gb2, out + 9 * (size_t)N, N);

    dim3 g2(Mt, ZCOLS / 32);
    wmma_gemm<HIDF, ZCOLS, true, true><<<g2, 32, 0, stream>>>(h, Wcat, bcat, zcat);
    k_heads<<<(N + 7) / 8, 256, 0, stream>>>(zcat, hw2, hb2, sw2, sb2, out, N);
}
